// MaNN2_43396349558997
// MI455X (gfx1250) — matrix operand order repaired and checked by reading, not compiled, not run on hardware
//
#include <hip/hip_runtime.h>

typedef __attribute__((ext_vector_type(16))) _Float16 v16h;
typedef __attribute__((ext_vector_type(2)))  _Float16 v2h;
typedef __attribute__((ext_vector_type(8)))  float    v8f;
typedef __attribute__((ext_vector_type(4)))  float    f4;

#define DD 20
#define NLAYERS 1000

// ws layout: [A tiles: NLAYERS*64 x v16h][C biases: NLAYERS*64 x v8f]
// index = l*64 + t*32 + lane
#define AWORDS ((size_t)NLAYERS * 64)

__device__ __forceinline__ float fsig(float x) {
#if __has_builtin(__builtin_amdgcn_tanhf)
  // sigmoid(x) = 0.5*tanh(x/2)+0.5 : 1 trans op (v_tanh_f32) instead of exp+rcp
  return __builtin_fmaf(0.5f, __builtin_amdgcn_tanhf(x * 0.5f), 0.5f);
#else
  float e = __builtin_amdgcn_exp2f(x * -1.44269504088896f);
  return __builtin_amdgcn_rcpf(1.0f + e);
#endif
}

// Build one 16x32 (f16, K-padded) A tile of W[l] for this lane.
// k order used here (contiguous, same as B): lanes 0-15: M=lane, K=0..15;
// lanes 16-31: M=lane-16, K=16..31.
__device__ __forceinline__ v16h load_wtile(const float* __restrict__ Wl,
                                           int row, bool loHalf) {
  v16h a;
#pragma unroll
  for (int i = 0; i < 16; ++i) a[i] = (_Float16)0.0f;
  if (row < DD) {
    const float* wr = Wl + row * DD;
    const int lowK = loHalf ? 0 : 16;
    f4 w0 = *reinterpret_cast<const f4*>(wr + lowK);
    a[0] = (_Float16)w0.x; a[1] = (_Float16)w0.y;
    a[2] = (_Float16)w0.z; a[3] = (_Float16)w0.w;
    if (loHalf) {
      f4 w1 = *reinterpret_cast<const f4*>(wr + 4);
      f4 w2 = *reinterpret_cast<const f4*>(wr + 8);
      f4 w3 = *reinterpret_cast<const f4*>(wr + 12);
      a[4] = (_Float16)w1.x; a[5] = (_Float16)w1.y;
      a[6] = (_Float16)w1.z; a[7] = (_Float16)w1.w;
      a[8]  = (_Float16)w2.x; a[9]  = (_Float16)w2.y;
      a[10] = (_Float16)w2.z; a[11] = (_Float16)w2.w;
      a[12] = (_Float16)w3.x; a[13] = (_Float16)w3.y;
      a[14] = (_Float16)w3.z; a[15] = (_Float16)w3.w;
    }
  }
  return a;
}

// Bias in C/D layout: lane holds feature (r + 8*laneHalf + 16*tile) in slot r.
__device__ __forceinline__ v8f load_bias(const float* __restrict__ bl, int fbase) {
  v8f c;
#pragma unroll
  for (int i = 0; i < 8; ++i) c[i] = 0.0f;
  if (fbase < DD) {
    f4 b0 = *reinterpret_cast<const f4*>(bl + fbase);
    c[0] = b0.x; c[1] = b0.y; c[2] = b0.z; c[3] = b0.w;
    if (fbase + 4 < DD) {
      f4 b1 = *reinterpret_cast<const f4*>(bl + fbase + 4);
      c[4] = b1.x; c[5] = b1.y; c[6] = b1.z; c[7] = b1.w;
    }
  }
  return c;
}

// x row -> B-matrix lane registers (K=features padded to 32).
__device__ __forceinline__ v16h load_xtile(const float* __restrict__ x,
                                           int rowIdx, bool loHalf) {
  v16h b;
#pragma unroll
  for (int i = 0; i < 16; ++i) b[i] = (_Float16)0.0f;
  const float* xr = x + (size_t)rowIdx * DD;
  if (loHalf) {
    f4 a0 = *reinterpret_cast<const f4*>(xr + 0);
    f4 a1 = *reinterpret_cast<const f4*>(xr + 4);
    f4 a2 = *reinterpret_cast<const f4*>(xr + 8);
    f4 a3 = *reinterpret_cast<const f4*>(xr + 12);
    float xv[16] = {a0.x, a0.y, a0.z, a0.w, a1.x, a1.y, a1.z, a1.w,
                    a2.x, a2.y, a2.z, a2.w, a3.x, a3.y, a3.z, a3.w};
#pragma unroll
    for (int i = 0; i < 16; ++i) b[i] = (_Float16)xv[i];
  } else {                              // K=16..19 valid, 20..31 pad
    f4 a0 = *reinterpret_cast<const f4*>(xr + 16);
    b[0] = (_Float16)a0.x; b[1] = (_Float16)a0.y;
    b[2] = (_Float16)a0.z; b[3] = (_Float16)a0.w;
  }
  return b;
}

// One-time layout conversion: W -> packed f16 A-register tiles, bias -> C tiles.
__global__ __launch_bounds__(256) void MaNN2_prep_kernel(
    const float* __restrict__ W, const float* __restrict__ bias,
    char* __restrict__ ws) {
  const int tid = blockIdx.x * blockDim.x + threadIdx.x;  // (l, t, lane)
  if (tid >= NLAYERS * 64) return;
  const int lane    = tid & 31;
  const int t       = (tid >> 5) & 1;
  const int l       = tid >> 6;
  const bool loHalf = lane < 16;
  const int col     = lane & 15;

  v16h a = load_wtile(W + (size_t)l * (DD * DD), t * 16 + col, loHalf);
  v8f  c = load_bias(bias + (size_t)l * DD, t * 16 + (loHalf ? 0 : 8));
  reinterpret_cast<v16h*>(ws)[tid] = a;
  reinterpret_cast<v8f*>(ws + AWORDS * sizeof(v16h))[tid] = c;
}

template <bool PREPPED>
__device__ __forceinline__ void fetch_layer(
    int l, int lane, int col, bool loHalf,
    const v16h* __restrict__ Ap, const v8f* __restrict__ Cp,
    const float* __restrict__ W, const float* __restrict__ bias,
    v16h& a1, v16h& a2, v8f& c1, v8f& c2) {
  if constexpr (PREPPED) {
    const int idx = l * 64 + lane;      // straight-line b128 loads, L2 hits
    a1 = Ap[idx];
    a2 = Ap[idx + 32];
    c1 = Cp[idx];
    c2 = Cp[idx + 32];
  } else {
    const float* Wl = W + (size_t)l * (DD * DD);
    const float* bl = bias + (size_t)l * DD;
    a1 = load_wtile(Wl, col, loHalf);
    a2 = load_wtile(Wl, 16 + col, loHalf);
    c1 = load_bias(bl, loHalf ? 0 : 8);
    c2 = load_bias(bl, loHalf ? 16 : 24);
  }
}

// sigmoid both feature tiles, pack to f16, exchange packed dwords with the
// xor-16 partner, and return the next-layer B matrix registers.
__device__ __forceinline__ v16h layer_post(v8f d1, v8f d2, bool loHalf,
                                           float* __restrict__ s1,
                                           float* __restrict__ s2) {
#pragma unroll
  for (int r = 0; r < 8; ++r) { s1[r] = fsig(d1[r]); s2[r] = fsig(d2[r]); }
  v2h p1[4], p2[4];
#pragma unroll
  for (int k = 0; k < 4; ++k) {
    p1[k] = __builtin_bit_cast(
        v2h, __builtin_amdgcn_cvt_pkrtz(s1[2 * k], s1[2 * k + 1]));
    p2[k] = __builtin_bit_cast(
        v2h, __builtin_amdgcn_cvt_pkrtz(s2[2 * k], s2[2 * k + 1]));
  }
  union { v2h p[8]; v16h v; } ub;
#pragma unroll
  for (int k = 0; k < 4; ++k) {
    v2h q1 = __builtin_bit_cast(
        v2h, __shfl_xor(__builtin_bit_cast(int, p1[k]), 16, 32));
    v2h q2 = __builtin_bit_cast(
        v2h, __shfl_xor(__builtin_bit_cast(int, p2[k]), 16, 32));
    ub.p[k]     = loHalf ? p1[k] : q2;  // K=0..7  / 16..23
    ub.p[k + 4] = loHalf ? q1 : p2[k];  // K=8..15 / 24..31
  }
  return ub.v;
}

template <bool PREPPED>
__global__ __launch_bounds__(256) void MaNN2_43396349558997_kernel(
    const float* __restrict__ x, const float* __restrict__ W,
    const float* __restrict__ bias, const float* __restrict__ Wout,
    const float* __restrict__ bout, float* __restrict__ out,
    const char* __restrict__ ws) {
  const int lane      = threadIdx.x & 31;
  const int wave      = (blockIdx.x * blockDim.x + threadIdx.x) >> 5;
  const int batchBase = wave * 32;     // two 16-row tiles per wave
  const int col       = lane & 15;     // N (batch) index inside tile
  const bool loHalf   = lane < 16;

  // two independent activation tiles (rows base..base+15, base+16..base+31)
  v16h bmatA = load_xtile(x, batchBase + col, loHalf);
  v16h bmatB = load_xtile(x, batchBase + 16 + col, loHalf);

  const v16h* __restrict__ Ap = reinterpret_cast<const v16h*>(ws);
  const v8f*  __restrict__ Cp =
      reinterpret_cast<const v8f*>(ws + AWORDS * sizeof(v16h));

  float sA1[8], sA2[8], sB1[8], sB2[8];

  // software pipeline: weights/bias for layer l are fetched in iteration l-1
  v16h a1, a2;
  v8f  c1, c2;
  fetch_layer<PREPPED>(0, lane, col, loHalf, Ap, Cp, W, bias, a1, a2, c1, c2);

  for (int l = 0; l < NLAYERS; ++l) {
    v16h a1n, a2n;
    v8f  c1n, c2n;
    const int ln = (l + 1 < NLAYERS) ? (l + 1) : l;
    fetch_layer<PREPPED>(ln, lane, col, loHalf, Ap, Cp, W, bias,
                         a1n, a2n, c1n, c2n);

    // D = W_tile x h^T + b : batch in lanes, features in VGPR slots.
    // 4 independent WMMAs (2 batch tiles x 2 feature tiles) share a1/a2/c1/c2.
    v8f dA1 = __builtin_amdgcn_wmma_f32_16x16x32_f16(false, a1, false, bmatA,
                                                     (short)0, c1, false, false);
    v8f dB1 = __builtin_amdgcn_wmma_f32_16x16x32_f16(false, a1, false, bmatB,
                                                     (short)0, c1, false, false);
    v8f dA2 = __builtin_amdgcn_wmma_f32_16x16x32_f16(false, a2, false, bmatA,
                                                     (short)0, c2, false, false);
    v8f dB2 = __builtin_amdgcn_wmma_f32_16x16x32_f16(false, a2, false, bmatB,
                                                     (short)0, c2, false, false);

    bmatA = layer_post(dA1, dA2, loHalf, sA1, sA2);
    bmatB = layer_post(dB1, dB2, loHalf, sB1, sB2);

    a1 = a1n; a2 = a2n; c1 = c1n; c2 = c2n;
  }

  // ---- final Linear(20,1) + sigmoid: per-lane partial dot, xor-16 reduce ----
  const int f1base = loHalf ? 0 : 8;    // s*1[r] = feature f1base + r
  const int f2base = loHalf ? 16 : 24;  // s*2[r] = feature f2base + r
  float accA = 0.0f, accB = 0.0f;
#pragma unroll
  for (int r = 0; r < 8; ++r) {
    float w1 = Wout[f1base + r];
    accA += sA1[r] * w1;
    accB += sB1[r] * w1;
    int f2 = f2base + r;
    if (f2 < DD) {
      float w2 = Wout[f2];
      accA += sA2[r] * w2;
      accB += sB2[r] * w2;
    }
  }
  accA += __shfl_xor(accA, 16, 32);
  accB += __shfl_xor(accB, 16, 32);
  const float b0 = bout[0];
  if (loHalf) {
    out[batchBase + col]      = fsig(accA + b0);
    out[batchBase + 16 + col] = fsig(accB + b0);
  }
}

extern "C" void kernel_launch(void* const* d_in, const int* in_sizes, int n_in,
                              void* d_out, int out_size, void* d_ws, size_t ws_size,
                              hipStream_t stream) {
  const float* x    = (const float*)d_in[0];
  const float* W    = (const float*)d_in[1];
  const float* b    = (const float*)d_in[2];
  const float* Wout = (const float*)d_in[3];
  const float* bout = (const float*)d_in[4];
  float* out = (float*)d_out;

  const int Bsz    = in_sizes[0] / DD;  // 16384 batch rows
  const int waves  = Bsz / 32;          // 32 rows per wave32 (two tiles)
  const int thr    = 256;               // 8 waves per block
  const int blocks = (waves * 32 + thr - 1) / thr;

  const size_t need = AWORDS * (sizeof(v16h) + sizeof(v8f));  // ~4 MB
  if (d_ws != nullptr && ws_size >= need) {
    const int pn = NLAYERS * 64;
    MaNN2_prep_kernel<<<(pn + thr - 1) / thr, thr, 0, stream>>>(
        W, b, (char*)d_ws);
    MaNN2_43396349558997_kernel<true><<<blocks, thr, 0, stream>>>(
        x, W, b, Wout, bout, out, (const char*)d_ws);
  } else {
    MaNN2_43396349558997_kernel<false><<<blocks, thr, 0, stream>>>(
        x, W, b, Wout, bout, out, nullptr);
  }
}
